// fc_hgnn_86139864089147
// MI455X (gfx1250) — matrix operand order repaired and checked by reading, not compiled, not run on hardware
//
#include <hip/hip_runtime.h>
#include <math.h>

// ---------------------------------------------------------------------------
// FC-HGNN forward for MI455X (gfx1250, wave32).
//  * GCN feature transforms run as bf16 WMMA GEMMs (v_wmma_f32_16x16x32_bf16),
//    fp32 inputs converted to bf16 while staging tiles in LDS, K zero-padded.
//    A fragments = 2x ds_load_b128 from row-major tile; B staged pre-swizzled
//    into fragment-major [K/16][N][16] so each lane reads one contiguous 32B run.
//    K-loop templated/unrolled; A fragments hoisted and reused across N chunks.
//  * Edge aggregation / degree norm / attention softmax use atomics (L2-resident:
//    each subject's 100x64 agg tile is ~25KB, total << 192MB L2).
// ---------------------------------------------------------------------------

typedef __attribute__((ext_vector_type(16))) __bf16 v16bf;
typedef __attribute__((ext_vector_type(8)))  __bf16 v8bf;
typedef __attribute__((ext_vector_type(8)))  float  v8f;

#define NPSC 100
#define HALF_NPSC 50

__device__ __forceinline__ float leakyf(float x) { return x > 0.f ? x : 0.01f * x; }
__device__ __forceinline__ bool node_left(long node) { return (node % NPSC) < HALF_NPSC; }

__device__ __forceinline__ void atomicMaxF(float* a, float v) {
  if (v >= 0.f) atomicMax((int*)a, __float_as_int(v));
  else          atomicMin((unsigned int*)a, __float_as_uint(v));
}

// ------------------------------ utility ------------------------------------
__global__ void k_zero_f32(float* p, long n) {
  long t = (long)blockIdx.x * blockDim.x + threadIdx.x;
  if (t < n) p[t] = 0.f;
}
__global__ void k_zero_bf16(__bf16* p, int n) {
  int t = blockIdx.x * blockDim.x + threadIdx.x;
  if (t < n) p[t] = (__bf16)0.f;
}
// pack fp32 W[Kact x Nact] into bf16 Bp[Kpad x Npad] at column colOff
__global__ void k_pack(const float* __restrict__ src, __bf16* __restrict__ dst,
                       int Kact, int Nact, int Npad, int colOff) {
  int t = blockIdx.x * blockDim.x + threadIdx.x;
  if (t >= Kact * Nact) return;
  int k = t / Nact, n = t % Nact;
  dst[k * Npad + colOff + n] = (__bf16)src[t];
}

// ------------------------- degrees / symmetric norm ------------------------
__global__ void k_degrees(const int* __restrict__ ei, const float* __restrict__ w,
                          long E, float* dl, float* dr, float* dg) {
  long e = (long)blockIdx.x * blockDim.x + threadIdx.x;
  if (e >= E) return;
  long s = ei[e], d = ei[E + e];
  float wv = w[e];
  bool ls = node_left(s), ld = node_left(d);
  if (ls && ld)   atomicAdd(&dl[d], wv);
  if (!ls && !ld) atomicAdd(&dr[d], wv);
  atomicAdd(&dg[d], wv);
}
__global__ void k_dinv(float* d, long n) {   // deg -> rsqrt(deg + 1)  (self loop)
  long t = (long)blockIdx.x * blockDim.x + threadIdx.x;
  if (t < n) d[t] = rsqrtf(d[t] + 1.0f);
}

// --------------------------- WMMA bf16 GEMM --------------------------------
// C[M x Npad] = A[M x Kact](fp32) * Bp[Kpad x Npad](bf16).  KPAD%32==0, Npad%16==0.
// Block = 256 thr = 8 waves; each wave owns a 16-row stripe of a 128-row tile.
// LDS: sA row-major [128][KPAD] bf16; sB fragment-major [KPAD/16][Npad][16] bf16.
template <int KPAD>
__global__ __launch_bounds__(256) void k_gemm_bf16(
    const float* __restrict__ A, long M, int Kact,
    const __bf16* __restrict__ Bp, int Npad, float* __restrict__ C) {
  extern __shared__ char smem[];
  __bf16* sA = (__bf16*)smem;            // [128][KPAD]
  __bf16* sB = sA + 128 * KPAD;          // [KPAD/16][Npad][16]
  const int tid = threadIdx.x;
  const long row0 = (long)blockIdx.x * 128;

  for (int idx = tid; idx < 128 * KPAD; idx += 256) {
    int r = idx / KPAD, k = idx % KPAD;
    long grow = row0 + r;
    float v = (grow < M && k < Kact) ? A[grow * (long)Kact + k] : 0.f;
    sA[idx] = (__bf16)v;
  }
  for (int idx = tid; idx < KPAD * Npad; idx += 256) {
    int k = idx / Npad, n = idx % Npad;
    sB[((k >> 4) * Npad + n) * 16 + (k & 15)] = Bp[idx];
  }
  __syncthreads();

  const int lane = tid & 31, wv = tid >> 5;
  const int lhalf = lane >> 4, lq = lane & 15;   // ISA 7.12.2 lane split
  const int wrow = wv * 16;

  // Hoist A fragments: lane (lhalf,lq) reads rows wrow+lq; per 32-K chunk the
  // fragment is two contiguous 8-element runs -> two b128 LDS loads each.
  constexpr int KC = KPAD / 32;
  v16bf afr[KC];
  {
    const v8bf* aptr = (const v8bf*)&sA[(wrow + lq) * KPAD];
#pragma unroll
    for (int kc = 0; kc < KC; ++kc) {
      v8bf lo = aptr[kc * 4 + lhalf * 2];        // K = kb + lhalf*16 .. +7
      v8bf hi = aptr[kc * 4 + lhalf * 2 + 1];    // K = kb + lhalf*16 + 8 .. +15
      afr[kc] = __builtin_shufflevector(lo, hi, 0, 1, 2, 3, 4, 5, 6, 7,
                                        8, 9, 10, 11, 12, 13, 14, 15);
    }
  }

  const bool fullTile = (row0 + 128 <= M);
  for (int nb = 0; nb < Npad; nb += 16) {
    v8f acc = {};
#pragma unroll
    for (int kc = 0; kc < KC; ++kc) {
      // B fragment: K run (kc*32 + lhalf*16 .. +15) at column nb+lq, contiguous.
      const v8bf* bptr =
          (const v8bf*)&sB[((kc * 2 + lhalf) * Npad + nb + lq) * 16];
      v8bf blo = bptr[0], bhi = bptr[1];
      v16bf bfrag = __builtin_shufflevector(blo, bhi, 0, 1, 2, 3, 4, 5, 6, 7,
                                            8, 9, 10, 11, 12, 13, 14, 15);
      acc = __builtin_amdgcn_wmma_f32_16x16x32_bf16(
          false, afr[kc], false, bfrag, (short)0, acc, false, false);
    }
    // C: vgpr r, lanes 0-15 -> M=r, lanes 16-31 -> M=r+8
    if (fullTile) {
#pragma unroll
      for (int r = 0; r < 8; ++r)
        C[(row0 + wrow + r + lhalf * 8) * (long)Npad + nb + lq] = acc[r];
    } else {
#pragma unroll
      for (int r = 0; r < 8; ++r) {
        long grow = row0 + wrow + r + lhalf * 8;
        if (grow < M) C[grow * (long)Npad + nb + lq] = acc[r];
      }
    }
  }
}

// --------------------------- GCN combine kernels ---------------------------
// self-loop term + bias for both hemisphere branches
__global__ void k_self2(const float* __restrict__ xw, int sx, int offL, int offR, int C,
                        const float* dl, const float* dr,
                        const float* bl, const float* br,
                        float* aggL, float* aggR, long N) {
  long t = (long)blockIdx.x * blockDim.x + threadIdx.x;
  if (t >= N * C) return;
  long i = t / C; int c = t % C;
  float il = dl[i], ir = dr[i];
  aggL[t] = il * il * xw[i * sx + offL + c] + bl[c];
  aggR[t] = ir * ir * xw[i * sx + offR + c] + br[c];
}
__global__ void k_scatter2(const int* __restrict__ ei, const float* __restrict__ w,
                           long E, int C, const float* __restrict__ xw, int sx,
                           int offL, int offR, const float* dl, const float* dr,
                           float* aggL, float* aggR) {
  long t = (long)blockIdx.x * blockDim.x + threadIdx.x;
  if (t >= E * C) return;
  long e = t / C; int c = t % C;
  long s = ei[e], d = ei[E + e];
  float wv = w[e];
  bool ls = node_left(s), ld = node_left(d);
  if (ls && ld) {
    float nl = dl[s] * wv * dl[d];
    atomicAdd(&aggL[d * (long)C + c], xw[s * (long)sx + offL + c] * nl);
  }
  if (!ls && !ld) {
    float nr = dr[s] * wv * dr[d];
    atomicAdd(&aggR[d * (long)C + c], xw[s * (long)sx + offR + c] * nr);
  }
}
__global__ void k_select(const float* aggL, const float* aggR, int C, float* h, long N) {
  long t = (long)blockIdx.x * blockDim.x + threadIdx.x;
  if (t >= N * C) return;
  long i = t / C;
  h[t] = leakyf(node_left(i) ? aggL[t] : aggR[t]);
}
__global__ void k_selfg(const float* __restrict__ xw, int sx, int C,
                        const float* dg, const float* bg, float* agg, long N) {
  long t = (long)blockIdx.x * blockDim.x + threadIdx.x;
  if (t >= N * C) return;
  long i = t / C; int c = t % C;
  float ig = dg[i];
  agg[t] = ig * ig * xw[i * sx + c] + bg[c];
}
__global__ void k_scatterg(const int* __restrict__ ei, const float* __restrict__ w,
                           long E, int C, const float* __restrict__ xw, int sx,
                           const float* dg, float* agg) {
  long t = (long)blockIdx.x * blockDim.x + threadIdx.x;
  if (t >= E * C) return;
  long e = t / C; int c = t % C;
  long s = ei[e], d = ei[E + e];
  float ng = dg[s] * w[e] * dg[d];
  atomicAdd(&agg[d * (long)C + c], xw[s * (long)sx + c] * ng);
}
__global__ void k_act(const float* agg, float* h, long n) {
  long t = (long)blockIdx.x * blockDim.x + threadIdx.x;
  if (t < n) h[t] = leakyf(agg[t]);
}
__global__ void k_pool(const float* __restrict__ h3, int C, float* emb, int S) {
  int t = blockIdx.x * blockDim.x + threadIdx.x;
  if (t >= S * C) return;
  int s = t / C, c = t % C;
  float acc = 0.f;
  for (int j = 0; j < NPSC; ++j) acc += h3[((long)s * NPSC + j) * C + c];
  emb[t] = acc * (1.0f / NPSC);
}

// ----------------------------- population HPG ------------------------------
__global__ void k_bias8(float* q, int S,
                        const float* b0, const float* b1, const float* b2, const float* b3,
                        const float* b4, const float* b5, const float* b6, const float* b7) {
  int t = blockIdx.x * blockDim.x + threadIdx.x;
  if (t >= S * 160) return;
  int j = t % 160, which = j / 20, c = j % 20;
  const float* bs[8] = {b0, b1, b2, b3, b4, b5, b6, b7};
  q[t] += bs[which][c];
}
__global__ void k_att_init(float* m, float* den, float* agg, int S, int C) {
  int t = blockIdx.x * blockDim.x + threadIdx.x;
  if (t >= S * C) return;
  int i = t / C, c = t % C;
  if (c == 0) { m[i] = -3.402823466e38f; den[i] = 0.f; }
  agg[t] = 0.f;
}
__global__ void k_att_score(const int* __restrict__ ei, int EP,
                            const float* __restrict__ qk, int QS, int qo, int ko, int C,
                            float* sc, float* m) {
  int e = blockIdx.x * blockDim.x + threadIdx.x;
  if (e >= EP) return;
  long s = ei[e], d = ei[EP + e];
  float acc = 0.f;
  for (int c = 0; c < C; ++c) acc += qk[d * QS + qo + c] * qk[s * QS + ko + c];
  acc *= rsqrtf((float)C);
  sc[e] = acc;
  atomicMaxF(&m[d], acc);
}
__global__ void k_att_exp(const int* __restrict__ ei, int EP, float* sc,
                          const float* m, float* den) {
  int e = blockIdx.x * blockDim.x + threadIdx.x;
  if (e >= EP) return;
  long d = ei[EP + e];
  float ex = __expf(sc[e] - m[d]);
  sc[e] = ex;
  atomicAdd(&den[d], ex);
}
__global__ void k_att_agg(const int* __restrict__ ei, int EP,
                          const float* __restrict__ qk, int QS, int vo, int C,
                          const float* sc, float* agg) {
  long t = (long)blockIdx.x * blockDim.x + threadIdx.x;
  if (t >= (long)EP * C) return;
  int e = t / C, c = t % C;
  long s = ei[e], d = ei[EP + e];
  atomicAdd(&agg[d * (long)C + c], qk[s * QS + vo + c] * sc[e]);
}
__global__ void k_att_final(const float* __restrict__ qk, int QS,
                            const float* agg1, const float* den1, int so1,
                            const float* agg2, const float* den2, int so2,
                            const float* w1, const float* w2, int l,
                            float* xsc, int S, int C) {
  int t = blockIdx.x * blockDim.x + threadIdx.x;
  if (t >= S * C) return;
  long i = t / C; int c = t % C;
  float d1 = den1[i], d2 = den2[i];
  float x1 = (d1 > 0.f ? agg1[i * C + c] / d1 : 0.f) + qk[i * QS + so1 + c];
  float x2 = (d2 > 0.f ? agg2[i * C + c] / d2 : 0.f) + qk[i * QS + so2 + c];
  float a = w1[l], b = w2[l], tot = a + b;
  xsc[t] = (a / tot) * x1 + (b / tot) * x2;
}
__global__ void k_bn_stats(const float* __restrict__ x, int S, int C,
                           float* mean, float* var) {
  __shared__ float s1[256], s2[256];
  int c = blockIdx.x, tid = threadIdx.x;
  float a = 0.f, b = 0.f;
  for (int i = tid; i < S; i += 256) {
    float v = x[(long)i * C + c];
    a += v; b += v * v;
  }
  s1[tid] = a; s2[tid] = b; __syncthreads();
  for (int s = 128; s > 0; s >>= 1) {
    if (tid < s) { s1[tid] += s1[tid + s]; s2[tid] += s2[tid + s]; }
    __syncthreads();
  }
  if (tid == 0) {
    float mu = s1[0] / S;
    mean[c] = mu;
    var[c] = s2[0] / S - mu * mu;
  }
}
__global__ void k_bn_apply(const float* __restrict__ x, int S, int C,
                           const float* mean, const float* var,
                           const float* g, const float* b,
                           float* xs, float* fc, int l, int FCC) {
  int t = blockIdx.x * blockDim.x + threadIdx.x;
  if (t >= S * C) return;
  int i = t / C, c = t % C;
  float y = (x[t] - mean[c]) * rsqrtf(var[c] + 1e-5f) * g[c] + b[c];
  y = leakyf(y);
  xs[t] = y;
  fc[(long)i * FCC + l * C + c] = y;
}
__global__ void k_out(const float* __restrict__ fc, int S,
                      const float* __restrict__ W, const float* __restrict__ b,
                      float* out) {
  int t = blockIdx.x * blockDim.x + threadIdx.x;
  if (t >= S * 2) return;
  int i = t / 2, cls = t % 2;
  float acc = b[cls];
  for (int j = 0; j < 80; ++j) acc += fc[(long)i * 80 + j] * W[j * 2 + cls];
  out[t] = acc;
}

// ------------------------------- host side ---------------------------------
static inline unsigned gcount(long n) { return (unsigned)((n + 255) / 256); }

extern "C" void kernel_launch(void* const* d_in, const int* in_sizes, int n_in,
                              void* d_out, int out_size, void* d_ws, size_t ws_size,
                              hipStream_t stream) {
  (void)n_in; (void)out_size; (void)ws_size;
  // --- input map (jax pytree flatten: dict keys alphabetical) ---
  const float* x        = (const float*)d_in[0];
  const float* edge_w   = (const float*)d_in[1];
  const float* gcn_g_W  = (const float*)d_in[2];
  const float* gcn_g_b  = (const float*)d_in[3];
  const float* gcn_l1_W = (const float*)d_in[4];
  const float* gcn_l1_b = (const float*)d_in[5];
  const float* gcn_l2_W = (const float*)d_in[6];
  const float* gcn_l2_b = (const float*)d_in[7];
  const float* gcn_r1_W = (const float*)d_in[8];
  const float* gcn_r1_b = (const float*)d_in[9];
  const float* gcn_r2_W = (const float*)d_in[10];
  const float* gcn_r2_b = (const float*)d_in[11];
  // hpg layers: base 12 + 18*l: [bn_b, bn_g, c1{Wk,Wq,Wskip,Wv,bk,bq,bskip,bv}, c2{...}]
  const float* out_W = (const float*)d_in[84];
  const float* out_b = (const float*)d_in[85];
  const float* hw1   = (const float*)d_in[86];
  const float* hw2   = (const float*)d_in[87];
  const int* edge_index = (const int*)d_in[88];
  const int* same_index = (const int*)d_in[90];
  const int* diff_index = (const int*)d_in[91];

  const long N = (long)in_sizes[0] / 100;   // 800000 nodes
  const long E = (long)in_sizes[1];         // 4M edges
  const int  S = (int)(N / NPSC);           // 8000 subjects
  const int  EP = in_sizes[90] / 2;         // 256000 population edges

  // --- workspace bump allocator (~1.2 GB of the 432 GiB card) ---
  char* wsb = (char*)d_ws;
  size_t off = 0;
  auto ALLOC = [&](size_t bytes) -> void* {
    void* p = wsb + off;
    off += (bytes + 255) & ~(size_t)255;
    return p;
  };
  float* degs = (float*)ALLOC((size_t)3 * N * 4);
  float *dl = degs, *dr = degs + N, *dg = degs + 2 * N;
  __bf16* Bp  = (__bf16*)ALLOC(128 * 128 * 2);
  float* xw   = (float*)ALLOC((size_t)N * 128 * 4);
  float* aggL = (float*)ALLOC((size_t)N * 64 * 4);
  float* aggR = (float*)ALLOC((size_t)N * 64 * 4);
  float* h1   = (float*)ALLOC((size_t)N * 64 * 4);
  float* h2   = (float*)ALLOC((size_t)N * 20 * 4);
  float* h3   = (float*)ALLOC((size_t)N * 20 * 4);
  float* xs   = (float*)ALLOC((size_t)S * 20 * 4);
  float* xsc  = (float*)ALLOC((size_t)S * 20 * 4);
  float* qkvs = (float*)ALLOC((size_t)S * 160 * 4);
  float* fc   = (float*)ALLOC((size_t)S * 80 * 4);
  float* sc1  = (float*)ALLOC((size_t)EP * 4);
  float* m1   = (float*)ALLOC((size_t)S * 4);
  float* den1 = (float*)ALLOC((size_t)S * 4);
  float* agg1 = (float*)ALLOC((size_t)S * 20 * 4);
  float* sc2  = (float*)ALLOC((size_t)EP * 4);
  float* m2   = (float*)ALLOC((size_t)S * 4);
  float* den2 = (float*)ALLOC((size_t)S * 4);
  float* agg2 = (float*)ALLOC((size_t)S * 20 * 4);
  float* meanb = (float*)ALLOC(20 * 4);
  float* varb  = (float*)ALLOC(20 * 4);

  // --- degrees + symmetric norm (incl. self loop) ---
  k_zero_f32<<<gcount(3 * N), 256, 0, stream>>>(degs, 3 * N);
  k_degrees<<<gcount(E), 256, 0, stream>>>(edge_index, edge_w, E, dl, dr, dg);
  k_dinv<<<gcount(3 * N), 256, 0, stream>>>(degs, 3 * N);

  // --- GCN layer 1: x[Nx100] @ [Wl1|Wr1] -> xw[Nx128] (bf16 WMMA) ---
  k_zero_bf16<<<gcount(128 * 128), 256, 0, stream>>>(Bp, 128 * 128);
  k_pack<<<gcount(100 * 64), 256, 0, stream>>>(gcn_l1_W, Bp, 100, 64, 128, 0);
  k_pack<<<gcount(100 * 64), 256, 0, stream>>>(gcn_r1_W, Bp, 100, 64, 128, 64);
  k_gemm_bf16<128><<<gcount(N * 2), 256, (128 * 128 + 128 * 128) * 2, stream>>>(
      x, N, 100, Bp, 128, xw);
  k_self2<<<gcount(N * 64), 256, 0, stream>>>(xw, 128, 0, 64, 64, dl, dr,
                                              gcn_l1_b, gcn_r1_b, aggL, aggR, N);
  k_scatter2<<<gcount(E * 64), 256, 0, stream>>>(edge_index, edge_w, E, 64, xw, 128,
                                                 0, 64, dl, dr, aggL, aggR);
  k_select<<<gcount(N * 64), 256, 0, stream>>>(aggL, aggR, 64, h1, N);

  // --- GCN layer 2: h1[Nx64] @ [Wl2|Wr2] -> xw[Nx48] ---
  k_zero_bf16<<<gcount(64 * 48), 256, 0, stream>>>(Bp, 64 * 48);
  k_pack<<<gcount(64 * 20), 256, 0, stream>>>(gcn_l2_W, Bp, 64, 20, 48, 0);
  k_pack<<<gcount(64 * 20), 256, 0, stream>>>(gcn_r2_W, Bp, 64, 20, 48, 20);
  k_gemm_bf16<64><<<gcount(N * 2), 256, (128 * 64 + 64 * 48) * 2, stream>>>(
      h1, N, 64, Bp, 48, xw);
  k_self2<<<gcount(N * 20), 256, 0, stream>>>(xw, 48, 0, 20, 20, dl, dr,
                                              gcn_l2_b, gcn_r2_b, aggL, aggR, N);
  k_scatter2<<<gcount(E * 20), 256, 0, stream>>>(edge_index, edge_w, E, 20, xw, 48,
                                                 0, 20, dl, dr, aggL, aggR);
  k_select<<<gcount(N * 20), 256, 0, stream>>>(aggL, aggR, 20, h2, N);

  // --- GCN layer 3 (global): h2[Nx20] @ Wg -> xw[Nx32] ---
  k_zero_bf16<<<gcount(32 * 32), 256, 0, stream>>>(Bp, 32 * 32);
  k_pack<<<gcount(20 * 20), 256, 0, stream>>>(gcn_g_W, Bp, 20, 20, 32, 0);
  k_gemm_bf16<32><<<gcount(N * 2), 256, (128 * 32 + 32 * 32) * 2, stream>>>(
      h2, N, 20, Bp, 32, xw);
  k_selfg<<<gcount(N * 20), 256, 0, stream>>>(xw, 32, 20, dg, gcn_g_b, aggL, N);
  k_scatterg<<<gcount(E * 20), 256, 0, stream>>>(edge_index, edge_w, E, 20, xw, 32,
                                                 dg, aggL);
  k_act<<<gcount(N * 20), 256, 0, stream>>>(aggL, h3, N * 20);

  // --- per-subject mean pooling -> xs[Sx20] ---
  k_pool<<<gcount((long)S * 20), 256, 0, stream>>>(h3, 20, xs, S);

  // --- population HPG: 4 transformer-conv layers ---
  for (int l = 0; l < 4; ++l) {
    const int base = 12 + 18 * l;
    const float* bn_b = (const float*)d_in[base + 0];
    const float* bn_g = (const float*)d_in[base + 1];
    const float *Wk1 = (const float*)d_in[base + 2],  *Wq1 = (const float*)d_in[base + 3];
    const float *Ws1 = (const float*)d_in[base + 4],  *Wv1 = (const float*)d_in[base + 5];
    const float *bk1 = (const float*)d_in[base + 6],  *bq1 = (const float*)d_in[base + 7];
    const float *bs1 = (const float*)d_in[base + 8],  *bv1 = (const float*)d_in[base + 9];
    const float *Wk2 = (const float*)d_in[base + 10], *Wq2 = (const float*)d_in[base + 11];
    const float *Ws2 = (const float*)d_in[base + 12], *Wv2 = (const float*)d_in[base + 13];
    const float *bk2 = (const float*)d_in[base + 14], *bq2 = (const float*)d_in[base + 15];
    const float *bs2 = (const float*)d_in[base + 16], *bv2 = (const float*)d_in[base + 17];

    // pack [q1|k1|v1|skip1|q2|k2|v2|skip2] -> Bp[32 x 160]
    k_zero_bf16<<<gcount(32 * 160), 256, 0, stream>>>(Bp, 32 * 160);
    k_pack<<<gcount(400), 256, 0, stream>>>(Wq1, Bp, 20, 20, 160, 0);
    k_pack<<<gcount(400), 256, 0, stream>>>(Wk1, Bp, 20, 20, 160, 20);
    k_pack<<<gcount(400), 256, 0, stream>>>(Wv1, Bp, 20, 20, 160, 40);
    k_pack<<<gcount(400), 256, 0, stream>>>(Ws1, Bp, 20, 20, 160, 60);
    k_pack<<<gcount(400), 256, 0, stream>>>(Wq2, Bp, 20, 20, 160, 80);
    k_pack<<<gcount(400), 256, 0, stream>>>(Wk2, Bp, 20, 20, 160, 100);
    k_pack<<<gcount(400), 256, 0, stream>>>(Wv2, Bp, 20, 20, 160, 120);
    k_pack<<<gcount(400), 256, 0, stream>>>(Ws2, Bp, 20, 20, 160, 140);
    k_gemm_bf16<32><<<gcount((long)S * 2), 256, (128 * 32 + 32 * 160) * 2, stream>>>(
        xs, S, 20, Bp, 160, qkvs);
    k_bias8<<<gcount((long)S * 160), 256, 0, stream>>>(qkvs, S, bq1, bk1, bv1, bs1,
                                                       bq2, bk2, bv2, bs2);
    // conv1 (same-site edges)
    k_att_init<<<gcount((long)S * 20), 256, 0, stream>>>(m1, den1, agg1, S, 20);
    k_att_score<<<gcount(EP), 256, 0, stream>>>(same_index, EP, qkvs, 160, 0, 20, 20, sc1, m1);
    k_att_exp<<<gcount(EP), 256, 0, stream>>>(same_index, EP, sc1, m1, den1);
    k_att_agg<<<gcount((long)EP * 20), 256, 0, stream>>>(same_index, EP, qkvs, 160, 40, 20, sc1, agg1);
    // conv2 (diff-site edges)
    k_att_init<<<gcount((long)S * 20), 256, 0, stream>>>(m2, den2, agg2, S, 20);
    k_att_score<<<gcount(EP), 256, 0, stream>>>(diff_index, EP, qkvs, 160, 80, 100, 20, sc2, m2);
    k_att_exp<<<gcount(EP), 256, 0, stream>>>(diff_index, EP, sc2, m2, den2);
    k_att_agg<<<gcount((long)EP * 20), 256, 0, stream>>>(diff_index, EP, qkvs, 160, 120, 20, sc2, agg2);
    // weighted combine + batch-norm + leaky; writes next xs and fc slice
    k_att_final<<<gcount((long)S * 20), 256, 0, stream>>>(qkvs, 160, agg1, den1, 60,
                                                          agg2, den2, 140, hw1, hw2, l,
                                                          xsc, S, 20);
    k_bn_stats<<<20, 256, 0, stream>>>(xsc, S, 20, meanb, varb);
    k_bn_apply<<<gcount((long)S * 20), 256, 0, stream>>>(xsc, S, 20, meanb, varb,
                                                         bn_g, bn_b, xs, fc, l, 80);
  }

  // --- final linear: fc[Sx80] @ out_W[80x2] + out_b ---
  k_out<<<gcount((long)S * 2), 256, 0, stream>>>(fc, S, out_W, out_b, (float*)d_out);
}
